// GCNEncoder_66958540144840
// MI455X (gfx1250) — compile-verified, matrix operand order checked by reading, not run on hardware
//
#include <hip/hip_runtime.h>

// GCN (4-layer) for MI455X / gfx1250, wave32.
// Dense transforms via V_WMMA_F32_16X16X32_BF16; graph aggregation via
// coalesced per-edge wave32 gather + global f32 atomics (L2-resident target).

#define F128 128

typedef __attribute__((ext_vector_type(16))) __bf16 v16bf;
typedef __attribute__((ext_vector_type(8)))  float  v8f;

__device__ __forceinline__ unsigned short f32_to_bf16_rne(float x) {
  unsigned int u = __float_as_uint(x);
  u += 0x7FFFu + ((u >> 16) & 1u);   // round-to-nearest-even
  return (unsigned short)(u >> 16);
}

// ---------------- degree / normalization ----------------

__global__ void init_deg_kernel(float* deg, int n) {
  int i = blockIdx.x * blockDim.x + threadIdx.x;
  if (i < n) deg[i] = 1.0f;          // self-loop contributes 1
}

__global__ void edge_deg_kernel(const int* __restrict__ ei, float* deg, int e) {
  int i = blockIdx.x * blockDim.x + threadIdx.x;
  if (i < e) atomicAdd(&deg[ei[e + i]], 1.0f);   // dst = ei[E + i]
}

__global__ void deg_to_dinv_kernel(float* deg, int n) {
  int i = blockIdx.x * blockDim.x + threadIdx.x;
  if (i < n) deg[i] = rsqrtf(deg[i]);            // deg >= 1 always
}

// ---------------- weight packing into WMMA B-fragment layout ----------------
// B fragment (bf16, 32xK16 tile): lane = {n = lane&15, ksub = (lane>>4)*8};
// half i (0..15) holds K = kt*32 + ksub + i + (i>=8 ? 8 : 0).
// Packed layout: [(kt*ntiles + nt)*32 + lane][16 halves], contiguous.

__global__ void pack_weight_bf16(const float* __restrict__ W,
                                 unsigned short* __restrict__ Wp,
                                 int K, int Nout) {
  int idx = blockIdx.x * blockDim.x + threadIdx.x;
  if (idx >= K * Nout) return;
  int ntiles = Nout >> 4;
  int i    = idx & 15;
  int lane = (idx >> 4) & 31;
  int tile = idx >> 9;
  int nt = tile % ntiles;
  int kt = tile / ntiles;
  int n    = lane & 15;
  int ksub = (lane >> 4) << 3;
  int k = kt * 32 + ksub + i + ((i >= 8) ? 8 : 0);
  Wp[idx] = f32_to_bf16_rne(W[k * Nout + nt * 16 + n]);
}

// ---------------- GEMM: C[N,Nout] = (relu?)(A[N,128]) @ W ----------------
// One wave per 16-row stripe; computes all NT 16-col tiles, K = 128 (4 k-tiles).

template <bool RELU, int NT>
__launch_bounds__(256)
__global__ void gemm_bf16_wmma(const float* __restrict__ A,
                               const unsigned short* __restrict__ Wp,
                               float* __restrict__ C, int nrows) {
  const int K = 128;
  const int Nout = NT * 16;
  int gtid = blockIdx.x * blockDim.x + threadIdx.x;
  int wave = gtid >> 5;
  int lane = threadIdx.x & 31;
  int row0 = wave * 16;
  if (row0 >= nrows) return;   // wave-uniform: EXEC stays all-ones for WMMA

  const v8f vzero = {0.f, 0.f, 0.f, 0.f, 0.f, 0.f, 0.f, 0.f};
  v8f acc[NT];
#pragma unroll
  for (int i = 0; i < NT; ++i) acc[i] = vzero;

  const int m    = lane & 15;          // row within tile
  const int ksub = (lane >> 4) << 3;   // K sub-offset per half-wave
  const float* arow = A + (size_t)(row0 + m) * K;
  const v16bf* bbase = (const v16bf*)Wp;

#pragma unroll
  for (int kt = 0; kt < 4; ++kt) {
    const int kb = kt * 32 + ksub;     // 32B-aligned (ksub, kt*32 multiples of 8)
    float4 a0 = *(const float4*)(arow + kb);
    float4 a1 = *(const float4*)(arow + kb + 4);
    float4 a2 = *(const float4*)(arow + kb + 16);
    float4 a3 = *(const float4*)(arow + kb + 20);
    if (RELU) {
      a0.x = fmaxf(a0.x, 0.f); a0.y = fmaxf(a0.y, 0.f);
      a0.z = fmaxf(a0.z, 0.f); a0.w = fmaxf(a0.w, 0.f);
      a1.x = fmaxf(a1.x, 0.f); a1.y = fmaxf(a1.y, 0.f);
      a1.z = fmaxf(a1.z, 0.f); a1.w = fmaxf(a1.w, 0.f);
      a2.x = fmaxf(a2.x, 0.f); a2.y = fmaxf(a2.y, 0.f);
      a2.z = fmaxf(a2.z, 0.f); a2.w = fmaxf(a2.w, 0.f);
      a3.x = fmaxf(a3.x, 0.f); a3.y = fmaxf(a3.y, 0.f);
      a3.z = fmaxf(a3.z, 0.f); a3.w = fmaxf(a3.w, 0.f);
    }
    // Native f32->bf16 converts (RNE); clang packs pairs into cvt_pk ops.
    v16bf av = { (__bf16)a0.x, (__bf16)a0.y, (__bf16)a0.z, (__bf16)a0.w,
                 (__bf16)a1.x, (__bf16)a1.y, (__bf16)a1.z, (__bf16)a1.w,
                 (__bf16)a2.x, (__bf16)a2.y, (__bf16)a2.z, (__bf16)a2.w,
                 (__bf16)a3.x, (__bf16)a3.y, (__bf16)a3.z, (__bf16)a3.w };
#pragma unroll
    for (int nt = 0; nt < NT; ++nt) {
      v16bf bf = bbase[((kt * NT + nt) << 5) + lane];
      acc[nt] = __builtin_amdgcn_wmma_f32_16x16x32_bf16(
          false, av, false, bf, (short)0, acc[nt], false, false);
    }
  }

  // C/D layout: VGPR r -> M = (lane<16 ? r : 8+r), N = lane&15
  const int n = lane & 15;
  const int rbase = (lane >> 4) << 3;
#pragma unroll
  for (int nt = 0; nt < NT; ++nt) {
#pragma unroll
    for (int r = 0; r < 8; ++r) {
      C[(size_t)(row0 + rbase + r) * Nout + nt * 16 + n] = acc[nt][r];
    }
  }
}

// ---------------- aggregation ----------------
// out[i,f] = bias[f] + Ht[i,f]*dinv[i]^2   (self-loop term; full init of out)

__global__ void init_agg_kernel(const float* __restrict__ Ht,
                                const float* __restrict__ dinv,
                                const float* __restrict__ bias,
                                float* __restrict__ out, int n, int F) {
  long long idx = (long long)blockIdx.x * blockDim.x + threadIdx.x;
  long long total = (long long)n * F;
  if (idx >= total) return;
  int i = (int)(idx / F);
  int f = (int)(idx % F);
  float di = dinv[i];
  out[idx] = bias[f] + Ht[idx] * di * di;
}

// One wave32 per edge (grid-stride): gather Ht[src]*norm, atomic-add to out[dst].
__global__ void edge_scatter_kernel(const float* __restrict__ Ht,
                                    const float* __restrict__ dinv,
                                    const int* __restrict__ ei,
                                    float* __restrict__ out, int e, int F) {
  int wid  = (blockIdx.x * blockDim.x + threadIdx.x) >> 5;
  int lane = threadIdx.x & 31;
  int nw   = (gridDim.x * blockDim.x) >> 5;
  for (int ed = wid; ed < e; ed += nw) {
    int s = ei[ed];
    int d = ei[e + ed];
    float w = dinv[s] * dinv[d];
    const float* hs = Ht + (size_t)s * F;
    float* od = out + (size_t)d * F;
    for (int f = lane; f < F; f += 32)
      atomicAdd(&od[f], hs[f] * w);
  }
}

// ---------------- launcher ----------------

extern "C" void kernel_launch(void* const* d_in, const int* in_sizes, int n_in,
                              void* d_out, int out_size, void* d_ws, size_t ws_size,
                              hipStream_t stream) {
  const float* x   = (const float*)d_in[0];
  const int*   ei  = (const int*)d_in[1];
  const float* W1  = (const float*)d_in[2];
  const float* b1  = (const float*)d_in[3];
  const float* Wm1 = (const float*)d_in[4];
  const float* bm1 = (const float*)d_in[5];
  const float* Wm2 = (const float*)d_in[6];
  const float* bm2 = (const float*)d_in[7];
  const float* W2  = (const float*)d_in[8];
  const float* b2  = (const float*)d_in[9];

  const int N = in_sizes[0] / F128;  // 100000 (multiple of 16)
  const int E = in_sizes[1] / 2;     // 1.6M
  float* out = (float*)d_out;        // [N, 64]

  // Workspace: dinv[N] | Htmp[N*128] | Hagg[N*128] | packed bf16 weights
  float* dinv = (float*)d_ws;
  float* Htmp = dinv + N;
  float* Hagg = Htmp + (size_t)N * F128;
  unsigned short* Wp1  = (unsigned short*)(Hagg + (size_t)N * F128);
  unsigned short* Wpm1 = Wp1  + 128 * 128;
  unsigned short* Wpm2 = Wpm1 + 128 * 128;
  unsigned short* Wp2  = Wpm2 + 128 * 128;

  const int T = 256;
  int bN  = (N + T - 1) / T;
  int bE  = (E + T - 1) / T;
  int bNF = (int)(((long long)N * F128 + T - 1) / T);
  int bNO = (int)(((long long)N * 64   + T - 1) / T);
  int waves  = (N + 15) / 16;
  int bGemm  = (waves * 32 + T - 1) / T;
  int bScat  = 4096;

  // 1) symmetric normalization
  init_deg_kernel   <<<bN, T, 0, stream>>>(dinv, N);
  edge_deg_kernel   <<<bE, T, 0, stream>>>(ei, dinv, E);
  deg_to_dinv_kernel<<<bN, T, 0, stream>>>(dinv, N);

  // 2) pack weights into WMMA B-fragment layout (bf16)
  pack_weight_bf16<<<(128 * 128 + T - 1) / T, T, 0, stream>>>(W1,  Wp1,  128, 128);
  pack_weight_bf16<<<(128 * 128 + T - 1) / T, T, 0, stream>>>(Wm1, Wpm1, 128, 128);
  pack_weight_bf16<<<(128 * 128 + T - 1) / T, T, 0, stream>>>(Wm2, Wpm2, 128, 128);
  pack_weight_bf16<<<(128 * 64  + T - 1) / T, T, 0, stream>>>(W2,  Wp2,  128, 64);

  // Layer 1: h = relu(agg(x @ W1) + b1)   (relu folded into next GEMM's load)
  gemm_bf16_wmma<false, 8><<<bGemm, T, 0, stream>>>(x, Wp1, Htmp, N);
  init_agg_kernel<<<bNF, T, 0, stream>>>(Htmp, dinv, b1, Hagg, N, F128);
  edge_scatter_kernel<<<bScat, T, 0, stream>>>(Htmp, dinv, ei, Hagg, E, F128);

  // Layer 2
  gemm_bf16_wmma<true, 8><<<bGemm, T, 0, stream>>>(Hagg, Wpm1, Htmp, N);
  init_agg_kernel<<<bNF, T, 0, stream>>>(Htmp, dinv, bm1, Hagg, N, F128);
  edge_scatter_kernel<<<bScat, T, 0, stream>>>(Htmp, dinv, ei, Hagg, E, F128);

  // Layer 3
  gemm_bf16_wmma<true, 8><<<bGemm, T, 0, stream>>>(Hagg, Wpm2, Htmp, N);
  init_agg_kernel<<<bNF, T, 0, stream>>>(Htmp, dinv, bm2, Hagg, N, F128);
  edge_scatter_kernel<<<bScat, T, 0, stream>>>(Htmp, dinv, ei, Hagg, E, F128);

  // Layer 4 (no relu): write directly to d_out [N, 64]
  gemm_bf16_wmma<true, 4><<<bGemm, T, 0, stream>>>(Hagg, Wp2, Htmp, N);
  init_agg_kernel<<<bNO, T, 0, stream>>>(Htmp, dinv, b2, out, N, 64);
  edge_scatter_kernel<<<bScat, T, 0, stream>>>(Htmp, dinv, ei, out, E, 64);
}
